// TF_MultiHeadAttention_9079560863776
// MI455X (gfx1250) — compile-verified, matrix operand order checked by reading, not run on hardware
//
#include <hip/hip_runtime.h>
#include <stdint.h>

// ---------------------------------------------------------------------------
// Types for CDNA5 WMMA (wave32): v16bf A/B fragments, v8f C/D fragments
// ---------------------------------------------------------------------------
typedef __bf16 bf16;
typedef __attribute__((ext_vector_type(16))) __bf16 v16bf;
typedef __attribute__((ext_vector_type(8)))  float  v8f;
typedef __attribute__((ext_vector_type(4)))  unsigned int v4u;
typedef __attribute__((ext_vector_type(8)))  unsigned int v8u;

#define B_  2
#define S_  2048
#define D_  1024
#define H_  16
#define KD_ 64

// ---------------------------------------------------------------------------
// Load a 16x32 bf16 A/B fragment from a row-major bf16 region.
// Per ISA 7.12.2 (16-bit A 16x32): lane l -> row = l&15, half = l>>4;
// VGPR 0..3 hold K = half*8 .. half*8+7 (16B contiguous),
// VGPR 4..7 hold K = 16+half*8 .. +7   (16B contiguous).
// B uses the mirrored layout (lane = N column, K contiguous per lane), so the
// caller stages B transposed ([N][K]) and reuses this loader.
// ---------------------------------------------------------------------------
__device__ __forceinline__ v16bf load_frag(const bf16* __restrict__ base, int ldm) {
    const int l = threadIdx.x & 31;
    const bf16* p = base + (l & 15) * ldm + (l >> 4) * 8;
    v16bf f;
#pragma unroll
    for (int i = 0; i < 8; ++i) f[i] = p[i];
#pragma unroll
    for (int i = 0; i < 8; ++i) f[i + 8] = p[16 + i];
    return f;
}

// ---------------------------------------------------------------------------
// Projection GEMM: dst[b,h,s,d] = (X[m,:] @ W[:,n]) + bias[n], n = h*64+d
// X: f32 [4096 x 1024], W: f32 [1024 x 1024]; staged to LDS as bf16.
// Block tile 64(M) x 128(N); 8 waves, each 16x64; K-step 32.
// ---------------------------------------------------------------------------
__global__ __launch_bounds__(256) void proj_gemm(const float* __restrict__ X,
                                                 const float* __restrict__ W,
                                                 const float* __restrict__ bias,
                                                 bf16* __restrict__ dst) {
    __shared__ bf16 lA[64 * 40];   // [m][k], row stride 40 (16B aligned, padded)
    __shared__ bf16 lB[128 * 40];  // transposed: [n][k]

    const int m0 = blockIdx.y * 64;
    const int n0 = blockIdx.x * 128;
    const int t = threadIdx.x;
    const int w = t >> 5, l = t & 31;
    const int wm = w >> 1, wn = w & 1;
    const int nl = l & 15, hl = l >> 4;

    v8f c[4] = {};

    for (int k0 = 0; k0 < 1024; k0 += 32) {
        // Stage A tile (64x32 f32 -> bf16): 8 contiguous elems/thread
        {
            const int idx = t * 8;
            const int row = idx >> 5, col = idx & 31;
            const float* g = X + (size_t)(m0 + row) * 1024 + k0 + col;
            if (k0 + 32 < 1024) __builtin_prefetch(g + 32, 0, 1);
            bf16* s = lA + row * 40 + col;
#pragma unroll
            for (int i = 0; i < 8; ++i) s[i] = (bf16)g[i];
        }
        // Stage B tile transposed (32x128 f32 -> bf16 [n][k])
#pragma unroll
        for (int kk = 0; kk < 2; ++kk) {
            const int k = kk * 16 + (t >> 4);
            const int nb = (t & 15) * 8;
            const float* g = W + (size_t)(k0 + k) * 1024 + n0 + nb;
            if (k0 + 32 < 1024) __builtin_prefetch(g + 32 * 1024, 0, 1);
#pragma unroll
            for (int i = 0; i < 8; ++i) lB[(nb + i) * 40 + k] = (bf16)g[i];
        }
        __syncthreads();

        const v16bf a = load_frag(lA + (wm * 16) * 40, 40);
#pragma unroll
        for (int j = 0; j < 4; ++j) {
            const v16bf bf = load_frag(lB + (wn * 64 + j * 16) * 40, 40);
            c[j] = __builtin_amdgcn_wmma_f32_16x16x32_bf16(false, a, false, bf,
                                                           (short)0, c[j], false, false);
        }
        __syncthreads();
    }

    // Epilogue: + bias, scatter to [B,H,S,64] bf16
#pragma unroll
    for (int j = 0; j < 4; ++j) {
        const int n = n0 + wn * 64 + j * 16 + nl;
        const float bv = bias[n];
        const int hh = n >> 6, d = n & 63;
#pragma unroll
        for (int r = 0; r < 8; ++r) {
            const int m = m0 + wm * 16 + r + 8 * hl;
            const int b = m >> 11, s = m & 2047;
            dst[(((size_t)b * H_ + hh) * S_ + s) * KD_ + d] = (bf16)(c[j][r] + bv);
        }
    }
}

// ---------------------------------------------------------------------------
// Flash-attention: one block = 128 q-rows of one (b,h); per wave 16 q-rows.
// Online softmax over K/V tiles of 32 rows. K tile is staged to LDS by the
// Tensor Data Mover (wave 0 builds D#, issues tensor_load_to_lds, waits
// TENSORcnt); V tile is transposed by hand; all matmuls via WMMA bf16.
// ---------------------------------------------------------------------------
__global__ __launch_bounds__(256) void attn_kernel(const bf16* __restrict__ qh,
                                                   const bf16* __restrict__ kh,
                                                   const bf16* __restrict__ vh,
                                                   const int* __restrict__ mask,
                                                   bf16* __restrict__ concat) {
    __shared__ bf16 lK[32 * 72];        // K tile [krow][d], stride 72 (144B = 128B + TDM pad)
    __shared__ bf16 lVt[64 * 40];       // V tile transposed [d][krow]
    __shared__ bf16 lP[8 * 16 * 40];    // per-wave P tile [qrow][krow]

    const int bh = blockIdx.y;          // b*H + h
    const int b = bh >> 4, hh = bh & 15;
    const int q0 = blockIdx.x * 128;
    const int t = threadIdx.x, w = t >> 5, l = t & 31;
    const int nl = l & 15, hl = l >> 4;
    const int qw = q0 + w * 16;

    const bf16* Qb = qh + ((size_t)bh * S_ + qw) * KD_;
    const bf16* Kb = kh + (size_t)bh * S_ * KD_;
    const bf16* Vb = vh + (size_t)bh * S_ * KD_;

    // Q fragments (d = 0..31 and 32..63), loaded once from global
    const v16bf qf0 = load_frag(Qb, KD_);
    const v16bf qf1 = load_frag(Qb + 32, KD_);

    v8f acc[4] = {};
    float mi[8], li[8];
#pragma unroll
    for (int r = 0; r < 8; ++r) { mi[r] = -1e30f; li[r] = 0.f; }

    const unsigned int lko = (unsigned int)(uintptr_t)(&lK[0]);  // LDS byte offset

    for (int k0 = 0; k0 < S_; k0 += 32) {
        // ---- K tile via Tensor Data Mover (wave 0 only; EXEC-independent) ----
        if (w == 0) {
            const unsigned long long ga =
                (unsigned long long)(uintptr_t)(Kb + (size_t)k0 * KD_);
            v4u g0;
            g0[0] = 1u;                                     // count=1 (valid user D#)
            g0[1] = lko;                                    // lds_addr (bytes)
            g0[2] = (unsigned int)ga;                       // global_addr[31:0]
            g0[3] = (unsigned int)(ga >> 32) | (2u << 30);  // global_addr[56:32] | type=2
            v8u g1;
            g1[0] = (1u << 16)      // data_size = 2 bytes
                  | (1u << 20)      // pad_enable
                  | (4u << 22)      // pad_interval: every 32 DWORDs (=128B row)
                  | (3u << 25);     // pad_amount: 4 DWORDs (=16B) -> 144B row stride
            g1[1] = (64u << 16);    // tensor_dim0 = 64 (bits 79:48, low half)
            g1[2] = (2048u << 16);  // tensor_dim0 hi | tensor_dim1 = 2048 (low half)
            g1[3] = (64u << 16);    // tensor_dim1 hi | tile_dim0 = 64
            g1[4] = 32u;            // tile_dim1 = 32 | tile_dim2 = 0
            g1[5] = 64u;            // tensor_dim0_stride = 64 (low 32)
            g1[6] = 0u;
            g1[7] = 0u;
            asm volatile("tensor_load_to_lds %0, %1" :: "s"(g0), "s"(g1) : "memory");
            __builtin_amdgcn_s_wait_tensorcnt(0);
        }
        // ---- V tile transposed (all threads) ----
        {
            const int r = t >> 3, db = (t & 7) * 8;
            const bf16* gv = Vb + (size_t)(k0 + r) * KD_ + db;
#pragma unroll
            for (int i = 0; i < 8; ++i) lVt[(db + i) * 40 + r] = gv[i];
        }
        __syncthreads();

        // scores (16 x 32) = Q (16x64) . K^T : two 16x16 subtiles, 2 d-steps each
        v8f sc0 = {}, sc1 = {};
        sc0 = __builtin_amdgcn_wmma_f32_16x16x32_bf16(false, qf0, false, load_frag(lK, 72),            (short)0, sc0, false, false);
        sc0 = __builtin_amdgcn_wmma_f32_16x16x32_bf16(false, qf1, false, load_frag(lK + 32, 72),       (short)0, sc0, false, false);
        sc1 = __builtin_amdgcn_wmma_f32_16x16x32_bf16(false, qf0, false, load_frag(lK + 16 * 72, 72),  (short)0, sc1, false, false);
        sc1 = __builtin_amdgcn_wmma_f32_16x16x32_bf16(false, qf1, false, load_frag(lK + 16 * 72 + 32, 72), (short)0, sc1, false, false);

        // scale 1/sqrt(64), apply mask
#pragma unroll
        for (int r = 0; r < 8; ++r) {
            const int qrow = qw + r + 8 * hl;
            const int mk0 = mask[(size_t)qrow * S_ + k0 + nl];
            const int mk1 = mask[(size_t)qrow * S_ + k0 + 16 + nl];
            sc0[r] = mk0 ? sc0[r] * 0.125f : -1e9f;
            sc1[r] = mk1 ? sc1[r] * 0.125f : -1e9f;
        }

        // row max across the 16 lanes of each half (rows live per-half)
        float rmax[8], rsum[8], corr[8];
#pragma unroll
        for (int r = 0; r < 8; ++r) rmax[r] = fmaxf(sc0[r], sc1[r]);
#pragma unroll
        for (int off = 1; off < 16; off <<= 1)
#pragma unroll
            for (int r = 0; r < 8; ++r) rmax[r] = fmaxf(rmax[r], __shfl_xor(rmax[r], off, 32));
#pragma unroll
        for (int r = 0; r < 8; ++r) {
            const float mn = fmaxf(mi[r], rmax[r]);
            corr[r] = __expf(mi[r] - mn);
            mi[r] = mn;
        }
#pragma unroll
        for (int r = 0; r < 8; ++r) {
            const float p0 = __expf(sc0[r] - mi[r]);
            const float p1 = __expf(sc1[r] - mi[r]);
            sc0[r] = p0; sc1[r] = p1; rsum[r] = p0 + p1;
        }
#pragma unroll
        for (int off = 1; off < 16; off <<= 1)
#pragma unroll
            for (int r = 0; r < 8; ++r) rsum[r] += __shfl_xor(rsum[r], off, 32);
#pragma unroll
        for (int r = 0; r < 8; ++r) li[r] = li[r] * corr[r] + rsum[r];
#pragma unroll
        for (int d = 0; d < 4; ++d)
#pragma unroll
            for (int r = 0; r < 8; ++r) acc[d][r] *= corr[r];

        // C-layout -> A-layout for P via per-wave LDS tile (same-wave RAW: dscnt)
        bf16* lPw = lP + w * 16 * 40;
#pragma unroll
        for (int r = 0; r < 8; ++r) {
            lPw[(r + 8 * hl) * 40 + nl]      = (bf16)sc0[r];
            lPw[(r + 8 * hl) * 40 + 16 + nl] = (bf16)sc1[r];
        }
        const v16bf pa = load_frag(lPw, 40);
#pragma unroll
        for (int d = 0; d < 4; ++d) {
            const v16bf bv = load_frag(lVt + d * 16 * 40, 40);
            acc[d] = __builtin_amdgcn_wmma_f32_16x16x32_bf16(false, pa, false, bv,
                                                             (short)0, acc[d], false, false);
        }
        __syncthreads();
    }

    // Normalize and write to concat [B, S, H*64] (bf16)
#pragma unroll
    for (int d = 0; d < 4; ++d)
#pragma unroll
        for (int r = 0; r < 8; ++r) {
            const int qrow = qw + r + 8 * hl;
            concat[((size_t)b * S_ + qrow) * (H_ * KD_) + hh * KD_ + d * 16 + nl] =
                (bf16)(acc[d][r] / li[r]);
        }
}

// ---------------------------------------------------------------------------
// Output GEMM: out[m,n] = concat[m,:](bf16) @ Wo[:,n](f32->bf16) + bo[n], f32 out
// A tile (already bf16) staged via async global->LDS engine (ASYNCcnt).
// ---------------------------------------------------------------------------
__global__ __launch_bounds__(256) void out_gemm(const bf16* __restrict__ A,
                                                const float* __restrict__ W,
                                                const float* __restrict__ bias,
                                                float* __restrict__ out) {
    __shared__ bf16 lA[64 * 40];
    __shared__ bf16 lB[128 * 40];

    const int m0 = blockIdx.y * 64;
    const int n0 = blockIdx.x * 128;
    const int t = threadIdx.x;
    const int w = t >> 5, l = t & 31;
    const int wm = w >> 1, wn = w & 1;
    const int nl = l & 15, hl = l >> 4;

    v8f c[4] = {};

    for (int k0 = 0; k0 < 1024; k0 += 32) {
        // A tile: one async 16B transfer per thread, direct to LDS (no VGPRs)
        {
            const int idx = t * 8;
            const int row = idx >> 5, col = idx & 31;
            const bf16* g = A + (size_t)(m0 + row) * 1024 + k0 + col;
            const unsigned int lo = (unsigned int)(uintptr_t)(lA + row * 40 + col);
            asm volatile("global_load_async_to_lds_b128 %0, %1, off"
                         :: "v"(lo), "v"(g) : "memory");
        }
        // B tile transposed (32x128 f32 -> bf16 [n][k])
#pragma unroll
        for (int kk = 0; kk < 2; ++kk) {
            const int k = kk * 16 + (t >> 4);
            const int nb = (t & 15) * 8;
            const float* g = W + (size_t)(k0 + k) * 1024 + n0 + nb;
            if (k0 + 32 < 1024) __builtin_prefetch(g + 32 * 1024, 0, 1);
#pragma unroll
            for (int i = 0; i < 8; ++i) lB[(nb + i) * 40 + k] = (bf16)g[i];
        }
        asm volatile("s_wait_asynccnt 0" ::: "memory");
        __syncthreads();

        const v16bf a = load_frag(lA + (wm * 16) * 40, 40);
#pragma unroll
        for (int j = 0; j < 4; ++j) {
            const v16bf bf = load_frag(lB + (wn * 64 + j * 16) * 40, 40);
            c[j] = __builtin_amdgcn_wmma_f32_16x16x32_bf16(false, a, false, bf,
                                                           (short)0, c[j], false, false);
        }
        __syncthreads();
    }

#pragma unroll
    for (int j = 0; j < 4; ++j) {
        const int n = n0 + wn * 64 + j * 16 + nl;
        const float bv = bias[n];
#pragma unroll
        for (int r = 0; r < 8; ++r) {
            const int m = m0 + wm * 16 + r + 8 * hl;
            out[(size_t)m * 1024 + n] = c[j][r] + bv;
        }
    }
}

// ---------------------------------------------------------------------------
// Host launch
// ---------------------------------------------------------------------------
extern "C" void kernel_launch(void* const* d_in, const int* in_sizes, int n_in,
                              void* d_out, int out_size, void* d_ws, size_t ws_size,
                              hipStream_t stream) {
    (void)in_sizes; (void)n_in; (void)out_size; (void)ws_size;
    const float* q    = (const float*)d_in[0];
    const float* k    = (const float*)d_in[1];
    const float* v    = (const float*)d_in[2];
    const int*   mask = (const int*)d_in[3];
    const float* Wq   = (const float*)d_in[4];
    const float* bq   = (const float*)d_in[5];
    const float* Wk   = (const float*)d_in[6];
    const float* bk   = (const float*)d_in[7];
    const float* Wv   = (const float*)d_in[8];
    const float* bv   = (const float*)d_in[9];
    const float* Wo   = (const float*)d_in[10];
    const float* bo   = (const float*)d_in[11];

    bf16* ws = (bf16*)d_ws;
    const size_t per = (size_t)B_ * H_ * S_ * KD_;  // 4,194,304 elems
    bf16* qhp = ws;
    bf16* khp = ws + per;
    bf16* vhp = ws + 2 * per;
    bf16* cc  = ws + 3 * per;

    const dim3 blk(256);
    const dim3 gemm_grid(8, 64);   // N/128 x M/64  (M = B*S = 4096, N = 1024)
    proj_gemm<<<gemm_grid, blk, 0, stream>>>(q, Wq, bq, qhp);
    proj_gemm<<<gemm_grid, blk, 0, stream>>>(k, Wk, bk, khp);
    proj_gemm<<<gemm_grid, blk, 0, stream>>>(v, Wv, bv, vhp);

    attn_kernel<<<dim3(S_ / 128, B_ * H_), blk, 0, stream>>>(qhp, khp, vhp, mask, cc);

    out_gemm<<<gemm_grid, blk, 0, stream>>>(cc, Wo, bo, (float*)d_out);
}
